// CPBlock_84370337563107
// MI455X (gfx1250) — matrix operand order repaired and checked by reading, not compiled, not run on hardware
//
#include <hip/hip_runtime.h>
#include <hip/hip_bf16.h>

// ---------------------------------------------------------------------------
// CDNA5 (gfx1250) implementation of the cross-view attention block.
// Heavy batched matmuls use v_wmma_f32_16x16x32_bf16 (wave32 WMMA).
// Per-pixel attention uses the rank-1-query algebraic fold (q has 1 row, so
// K/V projections collapse into small per-pixel VALU reductions).
// ---------------------------------------------------------------------------

#define CDIM   320
#define HEADS  10
#define DHEAD  32
#define LROWS  63          // (m-1)*9
#define NPIX   2048        // b*m*h*w
#define IMGS   8
#define HW     256         // 16*16

typedef __attribute__((ext_vector_type(16))) __bf16 v16bf;
typedef __attribute__((ext_vector_type(8)))  float  v8f;

__device__ __forceinline__ __bf16 f2bf(float x) { return (__bf16)x; }

union FragU { uint4 u[2]; v16bf v; };

// A fragment (16x32, 16-bit): lane M = lane&15; K = 16*(j>>3) + 8*(lane>>4) + (j&7)
__device__ __forceinline__ v16bf load_fragA(const __bf16* __restrict__ base, int ld,
                                            int r0, int k0, int lane) {
  const __bf16* p = base + (size_t)(r0 + (lane & 15)) * ld + k0 + 8 * ((lane >> 4) & 1);
  FragU f;
  f.u[0] = *(const uint4*)(p);
  f.u[1] = *(const uint4*)(p + 16);
  return f.v;
}

// B fragment (32x16, 16-bit): lane N = lane&15; K = 16*(j>>3) + 8*(lane>>4) + (j&7)
__device__ __forceinline__ v16bf load_fragB(const __bf16* __restrict__ base, int ld,
                                            int n0, int k0, int lane) {
  const __bf16* p = base + (size_t)(n0 + (lane & 15)) * ld + k0 + 8 * ((lane >> 4) & 1);
  FragU f;
  f.u[0] = *(const uint4*)(p);
  f.u[1] = *(const uint4*)(p + 16);
  return f.v;
}

// ---------------------------------------------------------------------------
// Generic WMMA GEMM:  C[M,N] = A[M,K] @ W[N,K]^T   (A,W bf16 row-major, K-contig)
// Fused epilogue: +bias, act(0=none,1=tanh-gelu), +resid (rowmajor or NCHW),
// store f32 rowmajor / bf16 rowmajor / f32 NCHW.
// Workgroup: 8 waves (256 thr) -> 128x64 tile; wave -> 32x32 (2x2 WMMA tiles).
// ---------------------------------------------------------------------------
__global__ __launch_bounds__(256) void gemm_bf16_kernel(
    const __bf16* __restrict__ A, const __bf16* __restrict__ W,
    int M, int N, int K,
    const float* __restrict__ bias, int act,
    const float* __restrict__ resid, int resid_nchw,
    float* __restrict__ outf, __bf16* __restrict__ outbf, int store_nchw)
{
  const int wave = threadIdx.x >> 5;
  const int lane = threadIdx.x & 31;
  const int wm = wave >> 1, wn = wave & 1;
  const int row0 = blockIdx.y * 128 + wm * 32;
  const int col0 = blockIdx.x * 64 + wn * 32;

  v8f acc[2][2] = {};

  for (int k0 = 0; k0 < K; k0 += 32) {
    v16bf a0 = load_fragA(A, K, row0,      k0, lane);
    v16bf a1 = load_fragA(A, K, row0 + 16, k0, lane);
    v16bf b0 = load_fragB(W, K, col0,      k0, lane);
    v16bf b1 = load_fragB(W, K, col0 + 16, k0, lane);
    acc[0][0] = __builtin_amdgcn_wmma_f32_16x16x32_bf16(false, a0, false, b0, (short)0, acc[0][0], false, false);
    acc[0][1] = __builtin_amdgcn_wmma_f32_16x16x32_bf16(false, a0, false, b1, (short)0, acc[0][1], false, false);
    acc[1][0] = __builtin_amdgcn_wmma_f32_16x16x32_bf16(false, a1, false, b0, (short)0, acc[1][0], false, false);
    acc[1][1] = __builtin_amdgcn_wmma_f32_16x16x32_bf16(false, a1, false, b1, (short)0, acc[1][1], false, false);
  }

  // C/D layout: M = r + 8*(lane>>4), N = lane&15
  for (int mi = 0; mi < 2; ++mi)
    for (int ni = 0; ni < 2; ++ni)
      for (int r = 0; r < 8; ++r) {
        int row = row0 + mi * 16 + r + 8 * (lane >> 4);
        int col = col0 + ni * 16 + (lane & 15);
        float v = acc[mi][ni][r];
        if (bias) v += bias[col];
        if (act == 1) {  // jax.nn.gelu default (approximate=True, tanh form)
          float x3 = v * v * v;
          v = 0.5f * v * (1.0f + tanhf(0.7978845608028654f * (v + 0.044715f * x3)));
        }
        size_t rm_idx   = (size_t)row * N + col;
        size_t nchw_idx = ((size_t)(row >> 8) * CDIM + col) * HW + (row & 255);
        if (resid) v += resid[resid_nchw ? nchw_idx : rm_idx];
        if (outbf)            outbf[rm_idx] = f2bf(v);
        else if (store_nchw)  outf[nchw_idx] = v;
        else                  outf[rm_idx]   = v;
      }
}

// ---------------------------------------------------------------------------
// Weight prep: transpose f32 (rows x cols) -> bf16 (cols x rows), and copy-convert
// ---------------------------------------------------------------------------
__global__ void transpose_bf16_kernel(const float* __restrict__ src, __bf16* __restrict__ dst,
                                      int rows, int cols) {
  size_t idx = (size_t)blockIdx.x * blockDim.x + threadIdx.x;
  if (idx >= (size_t)rows * cols) return;
  int i = (int)(idx / cols), j = (int)(idx % cols);
  dst[(size_t)j * rows + i] = f2bf(src[idx]);
}

__global__ void convert_bf16_kernel(const float* __restrict__ src, __bf16* __restrict__ dst, int n) {
  int idx = blockIdx.x * blockDim.x + threadIdx.x;
  if (idx < n) dst[idx] = f2bf(src[idx]);
}

// ---------------------------------------------------------------------------
// Relative SE(3) poses: PR[i,nn] = inv(P_i) @ P_j  (j = nn-th neighbor of i)
// ---------------------------------------------------------------------------
__global__ void prep_pose_kernel(const float* __restrict__ poses, float* __restrict__ PR) {
  int t = threadIdx.x;
  if (t >= IMGS * 7) return;
  int i = t / 7, nn = t % 7;
  int j = (nn < i) ? nn : nn + 1;
  const float* Pi = poses + i * 16;
  const float* Pj = poses + j * 16;
  float inv[16];
  for (int r = 0; r < 3; ++r)
    for (int c = 0; c < 3; ++c) inv[r * 4 + c] = Pi[c * 4 + r];  // R^T
  for (int r = 0; r < 3; ++r)
    inv[r * 4 + 3] = -(inv[r * 4 + 0] * Pi[3] + inv[r * 4 + 1] * Pi[7] + inv[r * 4 + 2] * Pi[11]);
  inv[12] = inv[13] = inv[14] = 0.f; inv[15] = 1.f;
  float* M = PR + t * 16;
  for (int r = 0; r < 4; ++r)
    for (int c = 0; c < 4; ++c) {
      float s = 0.f;
      for (int k = 0; k < 4; ++k) s += inv[r * 4 + k] * Pj[k * 4 + c];
      M[r * 4 + c] = s;
    }
}

// ---------------------------------------------------------------------------
// LayerNorm over channels: optional NCHW gather, optional +qpe, optional raw copy.
// One block per pixel row (320 channels), 256 threads.
// ---------------------------------------------------------------------------
__global__ __launch_bounds__(256) void ln_kernel(
    const float* __restrict__ in, int in_nchw,
    const float* __restrict__ g, const float* __restrict__ b,
    __bf16* __restrict__ outbf, float* __restrict__ copyf, int add_qpe)
{
  __shared__ float r1[256], r2[256];
  const int n = blockIdx.x, img = n >> 8, pix = n & 255;
  const int c0 = threadIdx.x, c1 = threadIdx.x + 256;
  float v0, v1 = 0.f;
  if (in_nchw) {
    v0 = in[((size_t)img * CDIM + c0) * HW + pix];
    if (c1 < CDIM) v1 = in[((size_t)img * CDIM + c1) * HW + pix];
  } else {
    v0 = in[(size_t)n * CDIM + c0];
    if (c1 < CDIM) v1 = in[(size_t)n * CDIM + c1];
  }
  r1[threadIdx.x] = v0 + v1;
  r2[threadIdx.x] = v0 * v0 + v1 * v1;
  __syncthreads();
  for (int st = 128; st > 0; st >>= 1) {
    if (threadIdx.x < st) {
      r1[threadIdx.x] += r1[threadIdx.x + st];
      r2[threadIdx.x] += r2[threadIdx.x + st];
    }
    __syncthreads();
  }
  const float mu  = r1[0] * (1.0f / CDIM);
  const float var = r2[0] * (1.0f / CDIM) - mu * mu;
  const float inv = rsqrtf(var + 1e-5f);
  {
    float hv = (v0 - mu) * inv * g[c0] + b[c0];
    if (add_qpe) hv += ((c0 / 80) & 1) ? 1.0f : 0.0f;   // posemb(0): sin->0, cos->1
    outbf[(size_t)n * CDIM + c0] = f2bf(hv);
    if (copyf) copyf[(size_t)n * CDIM + c0] = v0;
  }
  if (c1 < CDIM) {
    float hv = (v1 - mu) * inv * g[c1] + b[c1];
    if (add_qpe) hv += ((c1 / 80) & 1) ? 1.0f : 0.0f;
    outbf[(size_t)n * CDIM + c1] = f2bf(hv);
    if (copyf) copyf[(size_t)n * CDIM + c1] = v1;
  }
}

// ---------------------------------------------------------------------------
// Fused per-pixel attention (rank-1-query fold).
// One block (8 waves) per query pixel. Builds kv tile (gather + posemb + mask)
// in LDS, then: t = Wk-fold(q), scores = kv*t, softmax, kvw = w*kv,
// ctx = kvw @ Wv -> bf16 row for the Wo WMMA GEMM.
// ---------------------------------------------------------------------------
__global__ __launch_bounds__(256) void attn_kernel(
    const float* __restrict__ x,        // (8,320,16,16)
    const float* __restrict__ depths,   // (1,8,16,16)
    const float* __restrict__ corr,     // (1,8,8,16,16,2)
    const float* __restrict__ Kmat,     // (1,3,3)
    const float* __restrict__ Wk,       // (320,320) row-major [cin][cout]
    const float* __restrict__ Wv,       // (320,320)
    const float* __restrict__ PR,       // (8,7,16) relative poses
    const float* __restrict__ Qbuf,     // (2048,320)
    __bf16* __restrict__ Ctx)           // (2048,320) bf16
{
  __shared__ __bf16 kvT [LROWS * CDIM];   // 40320 B
  __shared__ __bf16 tmat[HEADS * CDIM];   //  6400 B
  __shared__ float  kvw [HEADS * CDIM];   // 12800 B
  __shared__ float  wsc [HEADS * 64];     //  2560 B
  __shared__ float  qrow[CDIM];           //  1280 B   (total 63360 B)

  const int tid  = threadIdx.x;
  const int wave = tid >> 5, lane = tid & 31;
  const int n    = blockIdx.x;
  const int img  = n >> 8, pix = n & 255;
  const int qy   = pix >> 4, qx = pix & 15;

  const float fx = Kmat[0], cx = Kmat[2], fyc = Kmat[4], cy = Kmat[5];
  const float depth_q = depths[img * HW + pix];

  // ---- build kv tile: 63 rows x 320 cols (gather + posemb + mask) ----
  for (int rr = 0; rr < 8; ++rr) {
    int l = wave * 8 + rr;
    if (l < LROWS) {
      int nn = l / 9, koff = l % 9;
      int j  = (nn < img) ? nn : nn + 1;
      int dxo = koff % 3 - 1, dyo = koff / 3 - 1;
      const float* cxy = corr + ((((size_t)img * IMGS + j) * 16 + qy) * 16 + qx) * 2;
      float xo = cxy[0] + (float)dxo, yo = cxy[1] + (float)dyo;
      float rxf = rintf(xo), ryf = rintf(yo);                    // round-half-even
      bool inb = (rxf >= 0.f) && (rxf < 16.f) && (ryf >= 0.f) && (ryf < 16.f);
      int rx = (int)fminf(fmaxf(rxf, 0.f), 15.f);
      int ry = (int)fminf(fmaxf(ryf, 0.f), 15.f);
      float dk = depths[j * HW + ry * 16 + rx];
      // back-project with K^-1, transform by relative pose, re-project with K
      float X = ((float)rx - cx) / fx * dk;
      float Y = ((float)ry - cy) / fyc * dk;
      float Z = dk;
      const float* M = PR + (img * 7 + nn) * 16;
      float px_ = M[0] * X + M[1] * Y + M[2]  * Z + M[3];
      float py_ = M[4] * X + M[5] * Y + M[6]  * Z + M[7];
      float pz_ = M[8] * X + M[9] * Y + M[10] * Z + M[11];
      float u = fx * px_ + cx * pz_;
      float v = fyc * py_ + cy * pz_;
      float zc = fmaxf(pz_, 1e-6f);
      float xpr = u / zc, ypr = v / zc;
      bool mask = inb && (dk > 1e-6f) && (pz_ > 1e-6f) &&
                  (fabsf(pz_ - depth_q) < 0.2f * depth_q);
      float xr = (xpr - (float)qx) * (1.0f / 16.0f);
      float yr = (ypr - (float)qy) * (1.0f / 16.0f);
      const float* featp = x + ((size_t)j * CDIM) * HW + ry * 16 + rx;
      const float lg = logf(10000.0f) / 80.0f;
      for (int c = lane; c < CDIM; c += 32) {
        float val = 0.f;
        if (mask) {
          int comp = c / 160;          // 0:x, 1:y
          int sc   = (c / 80) & 1;     // 0:sin, 1:cos
          int f    = c % 80;
          float fr  = __expf(-(float)f * lg);
          float ang = (comp ? yr : xr) * fr;
          float pe  = sc ? __cosf(ang) : __sinf(ang);
          val = featp[(size_t)c * HW] + pe;
        }
        kvT[l * CDIM + c] = f2bf(val);
      }
    }
  }
  for (int c = tid; c < CDIM; c += 256) qrow[c] = Qbuf[(size_t)n * CDIM + c];
  __syncthreads();

  // ---- t[h][c] = sum_d q[h*32+d] * Wk[c][h*32+d]  (fold K-projection) ----
  for (int hc = tid; hc < HEADS * CDIM; hc += 256) {
    int h = hc / CDIM, c = hc % CDIM;
    const float* wkr = Wk + (size_t)c * CDIM + h * DHEAD;
    float s = 0.f;
    for (int d = 0; d < DHEAD; ++d) s += qrow[h * DHEAD + d] * wkr[d];
    tmat[hc] = f2bf(s);
  }
  __syncthreads();

  // ---- scores[h][l] = (1/sqrt(32)) * kv[l] . t[h] ----
  for (int hl = tid; hl < HEADS * 64; hl += 256) {
    int h = hl >> 6, l = hl & 63;
    if (l < LROWS) {
      float s = 0.f;
      for (int c = 0; c < CDIM; ++c)
        s += (float)kvT[l * CDIM + c] * (float)tmat[h * CDIM + c];
      wsc[h * 64 + l] = s * 0.17677669529663687f;
    }
  }
  __syncthreads();

  // ---- softmax over l (63), per head ----
  if (tid < HEADS) {
    float mx = -1e30f;
    for (int l = 0; l < LROWS; ++l) mx = fmaxf(mx, wsc[tid * 64 + l]);
    float sum = 0.f;
    for (int l = 0; l < LROWS; ++l) {
      float e = __expf(wsc[tid * 64 + l] - mx);
      wsc[tid * 64 + l] = e; sum += e;
    }
    float inv = 1.0f / sum;
    for (int l = 0; l < LROWS; ++l) wsc[tid * 64 + l] *= inv;
  }
  __syncthreads();

  // ---- kvw[h][c] = sum_l w[h][l] * kv[l][c] ----
  for (int hc = tid; hc < HEADS * CDIM; hc += 256) {
    int h = hc / CDIM, c = hc % CDIM;
    float s = 0.f;
    for (int l = 0; l < LROWS; ++l) s += wsc[h * 64 + l] * (float)kvT[l * CDIM + c];
    kvw[hc] = s;
  }
  __syncthreads();

  // ---- ctx[c2] = sum_c kvw[head(c2)][c] * Wv[c][c2]  (fold V-projection) ----
  for (int c2 = tid; c2 < CDIM; c2 += 256) {
    int h = c2 / DHEAD;
    const float* kw = kvw + h * CDIM;
    const float* wv = Wv + c2;
    float s = 0.f;
    for (int c = 0; c < CDIM; ++c) s += kw[c] * wv[(size_t)c * CDIM];
    Ctx[(size_t)n * CDIM + c2] = f2bf(s);
  }
}

// ---------------------------------------------------------------------------
// GroupNorm(32 groups) + SiLU, NCHW in/out. One block per (img, group).
// ---------------------------------------------------------------------------
__global__ __launch_bounds__(256) void gn_silu_kernel(
    const float* __restrict__ in, const float* __restrict__ g,
    const float* __restrict__ b, float* __restrict__ out)
{
  __shared__ float r1[256], r2[256];
  const int img = blockIdx.x >> 5, grp = blockIdx.x & 31;
  const int CH = CDIM / 32;                 // 10 channels per group
  const size_t base = ((size_t)img * CDIM + grp * CH) * HW;
  float s = 0.f, s2 = 0.f;
  for (int i = threadIdx.x; i < CH * HW; i += 256) {
    float v = in[base + i]; s += v; s2 += v * v;
  }
  r1[threadIdx.x] = s; r2[threadIdx.x] = s2;
  __syncthreads();
  for (int st = 128; st > 0; st >>= 1) {
    if (threadIdx.x < st) {
      r1[threadIdx.x] += r1[threadIdx.x + st];
      r2[threadIdx.x] += r2[threadIdx.x + st];
    }
    __syncthreads();
  }
  const float cnt = (float)(CH * HW);
  const float mu  = r1[0] / cnt;
  const float var = r2[0] / cnt - mu * mu;
  const float inv = rsqrtf(var + 1e-5f);
  for (int i = threadIdx.x; i < CH * HW; i += 256) {
    int ch = grp * CH + i / HW;
    float v = (in[base + i] - mu) * inv * g[ch] + b[ch];
    out[base + i] = v / (1.0f + __expf(-v));   // silu
  }
}

// ---------------------------------------------------------------------------
// im2col for 3x3 SAME conv: Y(NCHW f32) -> A2[n][cin*9+ky*3+kx] bf16 (2048x2880)
// ---------------------------------------------------------------------------
__global__ void im2col_kernel(const float* __restrict__ Y, __bf16* __restrict__ A2) {
  size_t idx = (size_t)blockIdx.x * blockDim.x + threadIdx.x;
  if (idx >= (size_t)NPIX * (CDIM * 9)) return;
  int n = (int)(idx / (CDIM * 9));
  int k = (int)(idx % (CDIM * 9));
  int img = n >> 8, pix = n & 255, y = pix >> 4, xx = pix & 15;
  int cin = k / 9, r = k % 9, ky = r / 3, kx = r % 3;
  int yy = y + ky - 1, xq = xx + kx - 1;
  float v = 0.f;
  if (yy >= 0 && yy < 16 && xq >= 0 && xq < 16)
    v = Y[((size_t)img * CDIM + cin) * HW + yy * 16 + xq];
  A2[idx] = f2bf(v);
}

// ---------------------------------------------------------------------------
// Host-side launch
// ---------------------------------------------------------------------------
extern "C" void kernel_launch(void* const* d_in, const int* in_sizes, int n_in,
                              void* d_out, int out_size, void* d_ws, size_t ws_size,
                              hipStream_t stream) {
  const float* x       = (const float*)d_in[0];
  const float* poses   = (const float*)d_in[1];
  const float* Kmat    = (const float*)d_in[2];
  const float* depths  = (const float*)d_in[3];
  const float* corr    = (const float*)d_in[4];
  const float* Wq      = (const float*)d_in[5];
  const float* Wk      = (const float*)d_in[6];
  const float* Wv      = (const float*)d_in[7];
  const float* Wo      = (const float*)d_in[8];
  const float* ln1_g   = (const float*)d_in[9];
  const float* ln1_b   = (const float*)d_in[10];
  const float* ln2_g   = (const float*)d_in[11];
  const float* ln2_b   = (const float*)d_in[12];
  const float* ff_w1   = (const float*)d_in[13];
  const float* ff_b1   = (const float*)d_in[14];
  const float* ff_w2   = (const float*)d_in[15];
  const float* ff_b2   = (const float*)d_in[16];
  const float* gn1_g   = (const float*)d_in[17];
  const float* gn1_b   = (const float*)d_in[18];
  const float* gn2_g   = (const float*)d_in[19];
  const float* gn2_b   = (const float*)d_in[20];
  const float* conv1_w = (const float*)d_in[21];
  const float* conv1_b = (const float*)d_in[22];
  const float* conv2_w = (const float*)d_in[23];
  const float* conv2_b = (const float*)d_in[24];

  // ---- workspace carve-out (256B aligned slabs) ----
  char* ws = (char*)d_ws;
  size_t off = 0;
  auto alloc = [&](size_t bytes) -> void* {
    void* p = ws + off;
    off = (off + bytes + 255) & ~(size_t)255;
    return p;
  };
  __bf16* WqT = (__bf16*)alloc((size_t)CDIM * CDIM * 2);
  __bf16* WoT = (__bf16*)alloc((size_t)CDIM * CDIM * 2);
  __bf16* W1T = (__bf16*)alloc((size_t)1280 * CDIM * 2);
  __bf16* W2T = (__bf16*)alloc((size_t)CDIM * 1280 * 2);
  __bf16* CW1 = (__bf16*)alloc((size_t)CDIM * 2880 * 2);
  __bf16* CW2 = (__bf16*)alloc((size_t)CDIM * 2880 * 2);
  float*  PR  = (float*) alloc((size_t)IMGS * 7 * 16 * 4);
  __bf16* Hq  = (__bf16*)alloc((size_t)NPIX * CDIM * 2);
  float*  Qf  = (float*) alloc((size_t)NPIX * CDIM * 4);
  float*  Qb  = (float*) alloc((size_t)NPIX * CDIM * 4);
  __bf16* Ctx = (__bf16*)alloc((size_t)NPIX * CDIM * 2);
  float*  Xq  = (float*) alloc((size_t)NPIX * CDIM * 4);
  __bf16* H2  = (__bf16*)alloc((size_t)NPIX * CDIM * 2);
  __bf16* F1  = (__bf16*)alloc((size_t)NPIX * 1280 * 2);
  float*  OutA= (float*) alloc((size_t)NPIX * CDIM * 4);
  float*  Yb  = (float*) alloc((size_t)NPIX * CDIM * 4);
  __bf16* A2  = (__bf16*)alloc((size_t)NPIX * 2880 * 2);
  float*  H1  = (float*) alloc((size_t)NPIX * CDIM * 4);

  // ---- 1. weight prep ----
  transpose_bf16_kernel<<<400, 256, 0, stream>>>(Wq, WqT, CDIM, CDIM);
  transpose_bf16_kernel<<<400, 256, 0, stream>>>(Wo, WoT, CDIM, CDIM);
  transpose_bf16_kernel<<<1600, 256, 0, stream>>>(ff_w1, W1T, CDIM, 1280);
  transpose_bf16_kernel<<<1600, 256, 0, stream>>>(ff_w2, W2T, 1280, CDIM);
  convert_bf16_kernel<<<3600, 256, 0, stream>>>(conv1_w, CW1, CDIM * 2880);
  convert_bf16_kernel<<<3600, 256, 0, stream>>>(conv2_w, CW2, CDIM * 2880);
  prep_pose_kernel<<<1, 64, 0, stream>>>(poses, PR);

  // ---- 2. LN1 + qpe -> Hq (bf16), raw qf copy ----
  ln_kernel<<<NPIX, 256, 0, stream>>>(x, 1, ln1_g, ln1_b, Hq, Qf, 1);

  // ---- 3. Q projection (WMMA) ----
  gemm_bf16_kernel<<<dim3(5, 16), 256, 0, stream>>>(
      Hq, WqT, NPIX, CDIM, CDIM, nullptr, 0, nullptr, 0, Qb, nullptr, 0);

  // ---- 4. fused per-pixel attention ----
  attn_kernel<<<NPIX, 256, 0, stream>>>(x, depths, corr, Kmat, Wk, Wv, PR, Qb, Ctx);

  // ---- 5. O projection + residual (WMMA) -> Xq ----
  gemm_bf16_kernel<<<dim3(5, 16), 256, 0, stream>>>(
      Ctx, WoT, NPIX, CDIM, CDIM, nullptr, 0, Qf, 0, Xq, nullptr, 0);

  // ---- 6. LN2 -> H2 ----
  ln_kernel<<<NPIX, 256, 0, stream>>>(Xq, 0, ln2_g, ln2_b, H2, nullptr, 0);

  // ---- 7. FF1 + gelu (WMMA, bf16 out) ----
  gemm_bf16_kernel<<<dim3(20, 16), 256, 0, stream>>>(
      H2, W1T, NPIX, 1280, CDIM, ff_b1, 1, nullptr, 0, nullptr, F1, 0);

  // ---- 8. FF2 + residual -> OutA (NCHW) ----
  gemm_bf16_kernel<<<dim3(5, 16), 256, 0, stream>>>(
      F1, W2T, NPIX, CDIM, 1280, ff_b2, 0, Xq, 0, OutA, nullptr, 1);

  // ---- 9-11. GN1+SiLU, im2col, conv1 (WMMA implicit GEMM) ----
  gn_silu_kernel<<<256, 256, 0, stream>>>(OutA, gn1_g, gn1_b, Yb);
  im2col_kernel<<<23040, 256, 0, stream>>>(Yb, A2);
  gemm_bf16_kernel<<<dim3(5, 16), 256, 0, stream>>>(
      A2, CW1, NPIX, CDIM, 2880, conv1_b, 0, nullptr, 0, H1, nullptr, 1);

  // ---- 12-14. GN2+SiLU, im2col, conv2 + residual -> d_out ----
  gn_silu_kernel<<<256, 256, 0, stream>>>(H1, gn2_g, gn2_b, Yb);
  im2col_kernel<<<23040, 256, 0, stream>>>(Yb, A2);
  gemm_bf16_kernel<<<dim3(5, 16), 256, 0, stream>>>(
      A2, CW2, NPIX, CDIM, 2880, conv2_b, 0, OutA, 1, (float*)d_out, nullptr, 1);
}
